// DeCoR_2018634629147
// MI455X (gfx1250) — compile-verified, matrix operand order checked by reading, not run on hardware
//
#include <hip/hip_runtime.h>
#include <hip/hip_bf16.h>

// ---------------------------------------------------------------------------
// Deformable Conv1d pipeline for MI455X (gfx1250, wave32, WMMA).
//   B=8, L=4096, C=256, K=3, DIL=1
// Pipeline:
//   1) pad_convert : x[B,L,C] f32 -> xpad[B,L+2,C] f16 (zero rows at edges)
//   2) repack_w    : w[C,C,3] -> A[256,768] f16 with k index = t*256+ci
//   3) gemm (WMMA) : o[B,L,C] = relu(W1[256,768] x Xcol[768,L])
//   4) conv2       : offsets[B,L,3] (tiny C->3 conv, scalar FMA)
//   5) interp_pack : Y[B,L,3,C] f16 = g0*x[u0] + g1*x[u0+1]
//   6) gemm (WMMA) : out[B,L,C] = Wf[256,768] x Ycol[768,L]
// GEMMs: v_wmma_f32_16x16x32_f16, TRIPLE-buffered LDS fed by
// global_load_async_to_lds_b128; "issue next, wait asynccnt<=2" keeps the
// newest DMA pair in flight across the barrier and the WMMAs.
// ---------------------------------------------------------------------------

typedef __attribute__((ext_vector_type(16))) _Float16 v16h;
typedef __attribute__((ext_vector_type(8)))  _Float16 v8h;
typedef __attribute__((ext_vector_type(8)))  float    v8f;
typedef __attribute__((ext_vector_type(4)))  float    v4f;

#define DC_B   8
#define DC_L   4096
#define DC_C   256
#define DC_KD  768        // reduction dim = 3*C
#define TILE_K 32
#define NSTEPS (DC_KD / TILE_K)   // 24
#define LDSROW 40         // padded LDS row stride in halves (80B)
#define LDSBUF (64 * LDSROW)      // halves per buffer (5120 B)

// Raw 32-bit LDS byte offset of a shared-memory pointer (generic -> as(3) -> int).
__device__ __forceinline__ unsigned lds_off_u32(const void* p) {
  return (unsigned)(unsigned long long)(__attribute__((address_space(3))) const void*)p;
}

// ---------------------------------------------------------------------------
// 1) Pad + convert x -> f16 with one zero row at each end of L
// ---------------------------------------------------------------------------
__global__ __launch_bounds__(256) void pad_convert(const float* __restrict__ x,
                                                   _Float16* __restrict__ xp) {
  size_t i = (size_t)blockIdx.x * 256 + threadIdx.x;
  if (i >= (size_t)DC_B * (DC_L + 2) * DC_C) return;
  int ci = (int)(i & (DC_C - 1));
  size_t rowg = i >> 8;
  int r = (int)(rowg % (DC_L + 2));
  int b = (int)(rowg / (DC_L + 2));
  float v = 0.0f;
  if (r >= 1 && r <= DC_L)
    v = x[((size_t)b * DC_L + (r - 1)) * DC_C + ci];
  xp[i] = (_Float16)v;
}

// ---------------------------------------------------------------------------
// 2) Repack conv weight [co][ci][t] -> [co][t*256+ci], f32 -> f16
// ---------------------------------------------------------------------------
__global__ __launch_bounds__(256) void repack_w(const float* __restrict__ w,
                                                _Float16* __restrict__ Ah) {
  int i = blockIdx.x * 256 + threadIdx.x;
  if (i >= DC_C * DC_KD) return;
  int co = i / DC_KD;
  int r  = i - co * DC_KD;
  int t  = r >> 8;
  int ci = r & (DC_C - 1);
  Ah[i] = (_Float16)w[(size_t)co * DC_KD + ci * 3 + t];
}

// ---------------------------------------------------------------------------
// 3/6) WMMA GEMM: Out[b,l,co] = act( A[256,768] . Bcol[:,l] )
//   A: row-major f16. B columns: 768 contiguous halves per column, column l ->
//   l+1 stride = colStrideHalves. Block = 64(co) x 64(l), 8 wave32s.
//   Triple-buffered LDS, async global->LDS staging, 2 wmma / wave / K-step.
// ---------------------------------------------------------------------------
__global__ __launch_bounds__(256) void gemm_f16_wmma(
    const _Float16* __restrict__ A,
    const _Float16* __restrict__ Bb,
    float* __restrict__ Out,
    int colStrideHalves, long long bStrideHalves, int doRelu) {
  __shared__ _Float16 Alds[3 * LDSBUF];
  __shared__ _Float16 Blds[3 * LDSBUF];

  const int b    = blockIdx.z;
  const int co0  = blockIdx.y * 64;
  const int l0   = blockIdx.x * 64;
  const int tid  = threadIdx.x;
  const int lane = tid & 31;
  const int wave = tid >> 5;

  const _Float16* Bcols = Bb + (long long)b * bStrideHalves +
                          (long long)l0 * colStrideHalves;

  // staging: thread -> (tile row, 8-half segment); 16 bytes per thread per tile
  const int scol = tid & 63;
  const int sseg = tid >> 6;            // 0..3

  const _Float16* aSrc = A + (size_t)(co0 + scol) * DC_KD + sseg * 8;
  const _Float16* bSrc = Bcols + (long long)scol * colStrideHalves + sseg * 8;
  const unsigned aDstBase = lds_off_u32(&Alds[scol * LDSROW + sseg * 8]);
  const unsigned bDstBase = lds_off_u32(&Blds[scol * LDSROW + sseg * 8]);

  // async DMA of one 64x32-half tile pair into LDS buffer `buf` (2 ops)
  auto stage = [&](int buf, int k0) {
    const unsigned bufByte = (unsigned)(buf * LDSBUF * 2);
    unsigned long long ga = (unsigned long long)(const void*)(aSrc + k0);
    unsigned long long gb = (unsigned long long)(const void*)(bSrc + k0);
    asm volatile("global_load_async_to_lds_b128 %0, %1, off"
                 :: "v"(aDstBase + bufByte), "v"(ga) : "memory");
    asm volatile("global_load_async_to_lds_b128 %0, %1, off"
                 :: "v"(bDstBase + bufByte), "v"(gb) : "memory");
  };

  // fragment gather (CDNA5 16-bit A/B layout): lane%16 = row, lane/16 -> K lo/hi
  const int frow = lane & 15;
  const int klo  = (lane >> 4) << 3;    // 0 or 8

  const int sa = wave & 3;              // co subtile 0..3
  const int sn = (wave >> 2) << 1;      // l subtiles {sn, sn+1}

  v8f c0 = {};
  v8f c1 = {};

  stage(0, 0);                          // prefill buffer 0
  int buf = 0, nxt = 1;

  for (int it = 0; it < NSTEPS; ++it) {
    if (it + 1 < NSTEPS) {
      // issue next tile's DMA first, then retire only the *previous* pair:
      // async loads complete in order, so asynccnt<=2 leaves the new pair in
      // flight across the barrier and this iteration's WMMAs.
      stage(nxt, (it + 1) * TILE_K);
      asm volatile("s_wait_asynccnt 0x2" ::: "memory");
    } else {
      asm volatile("s_wait_asynccnt 0x0" ::: "memory");
    }
    __syncthreads();

    const _Float16* ap = &Alds[buf * LDSBUF + (sa * 16 + frow) * LDSROW + klo];
    v8h alo = *(const v8h*)(ap);
    v8h ahi = *(const v8h*)(ap + 16);
    v16h afrag = __builtin_shufflevector(alo, ahi, 0, 1, 2, 3, 4, 5, 6, 7,
                                         8, 9, 10, 11, 12, 13, 14, 15);

    const _Float16* bp0 = &Blds[buf * LDSBUF + ((sn + 0) * 16 + frow) * LDSROW + klo];
    v8h b0lo = *(const v8h*)(bp0);
    v8h b0hi = *(const v8h*)(bp0 + 16);
    v16h bfrag0 = __builtin_shufflevector(b0lo, b0hi, 0, 1, 2, 3, 4, 5, 6, 7,
                                          8, 9, 10, 11, 12, 13, 14, 15);

    const _Float16* bp1 = &Blds[buf * LDSBUF + ((sn + 1) * 16 + frow) * LDSROW + klo];
    v8h b1lo = *(const v8h*)(bp1);
    v8h b1hi = *(const v8h*)(bp1 + 16);
    v16h bfrag1 = __builtin_shufflevector(b1lo, b1hi, 0, 1, 2, 3, 4, 5, 6, 7,
                                          8, 9, 10, 11, 12, 13, 14, 15);

    c0 = __builtin_amdgcn_wmma_f32_16x16x32_f16(false, afrag, false, bfrag0,
                                                (short)0, c0, false, false);
    c1 = __builtin_amdgcn_wmma_f32_16x16x32_f16(false, afrag, false, bfrag1,
                                                (short)0, c1, false, false);

    // Buffer-reuse safety: buffer written at top of iter i was last read in
    // iter i-2; those ds_loads retired before that wave's barrier signal in
    // iter i-1 (s_wait_dscnt precedes the wmma), so 3 buffers + 1 barrier/iter
    // is race-free.
    buf = nxt;
    nxt = (nxt + 1 == 3) ? 0 : nxt + 1;
  }

  if (doRelu) {
#pragma unroll
    for (int i = 0; i < 8; ++i) {
      c0[i] = fmaxf(c0[i], 0.0f);
      c1[i] = fmaxf(c1[i], 0.0f);
    }
  }

  // C/D layout: VGPR r -> (M = r + 8*(lane/16), N = lane%16); per lane the 8
  // accumulator regs are 8 consecutive co -> two b128 stores into [B,L,C].
  const int n  = lane & 15;
  const int mb = (lane >> 4) << 3;
  const int coBase = co0 + sa * 16 + mb;
  {
    const int l = l0 + (sn + 0) * 16 + n;
    float* dst = Out + ((size_t)b * DC_L + l) * DC_C + coBase;
    v4f lo = {c0[0], c0[1], c0[2], c0[3]};
    v4f hi = {c0[4], c0[5], c0[6], c0[7]};
    *(v4f*)dst = lo;
    *(v4f*)(dst + 4) = hi;
  }
  {
    const int l = l0 + (sn + 1) * 16 + n;
    float* dst = Out + ((size_t)b * DC_L + l) * DC_C + coBase;
    v4f lo = {c1[0], c1[1], c1[2], c1[3]};
    v4f hi = {c1[4], c1[5], c1[6], c1[7]};
    *(v4f*)dst = lo;
    *(v4f*)(dst + 4) = hi;
  }
}

// ---------------------------------------------------------------------------
// 4) Offset conv: off[b,l,k] = sum_{ci,t} w2[k,ci,t] * o[b,l-1+t,ci]
// ---------------------------------------------------------------------------
__global__ __launch_bounds__(256) void conv2_offsets(const float* __restrict__ o,
                                                     const float* __restrict__ w2,
                                                     float* __restrict__ offb) {
  int idx = blockIdx.x * 256 + threadIdx.x;   // b*L + l
  if (idx >= DC_B * DC_L) return;
  int b = idx >> 12;
  int l = idx & (DC_L - 1);
  float s0 = 0.f, s1 = 0.f, s2 = 0.f;
#pragma unroll
  for (int t = 0; t < 3; ++t) {
    int m = l - 1 + t;
    if (m < 0 || m >= DC_L) continue;
    const float* orow = o + ((size_t)b * DC_L + m) * DC_C;
    for (int ci = 0; ci < DC_C; ++ci) {
      float v = orow[ci];
      s0 = fmaf(w2[0 * DC_KD + ci * 3 + t], v, s0);
      s1 = fmaf(w2[1 * DC_KD + ci * 3 + t], v, s1);
      s2 = fmaf(w2[2 * DC_KD + ci * 3 + t], v, s2);
    }
  }
  float* d = offb + (size_t)idx * 3;
  d[0] = s0;
  d[1] = s1;
  d[2] = s2;
}

// ---------------------------------------------------------------------------
// 5) Linear interp gather + f16 pack: Y[b,l,k,ci]
// ---------------------------------------------------------------------------
__global__ __launch_bounds__(256) void interp_pack(const float* __restrict__ x,
                                                   const float* __restrict__ offb,
                                                   _Float16* __restrict__ Yh) {
  int bl = blockIdx.x;                 // b*L + l
  int ci = threadIdx.x;
  int b = bl >> 12;
  int l = bl & (DC_L - 1);
  const float* ob = offb + (size_t)bl * 3;
  _Float16* yrow = Yh + (size_t)bl * DC_KD;
#pragma unroll
  for (int k = 0; k < 3; ++k) {
    float T = (float)l + (float)k + ob[k];
    T = fminf(fmaxf(T, (float)l), (float)(l + 2));    // clip to receptive field
    int u0 = (int)floorf(T);
    u0 = u0 < 0 ? 0 : (u0 > DC_L - 2 ? DC_L - 2 : u0);
    float uf = (float)u0;
    float g0 = fmaxf(1.0f - fabsf(uf - T), 0.0f);
    float g1 = fmaxf(1.0f - fabsf(uf + 1.0f - T), 0.0f);
    const float* x0 = x + ((size_t)b * DC_L + u0) * DC_C;
    float v = g0 * x0[ci] + g1 * x0[DC_C + ci];
    yrow[k * DC_C + ci] = (_Float16)v;
  }
}

// ---------------------------------------------------------------------------
// Host launch
// ---------------------------------------------------------------------------
extern "C" void kernel_launch(void* const* d_in, const int* in_sizes, int n_in,
                              void* d_out, int out_size, void* d_ws, size_t ws_size,
                              hipStream_t stream) {
  const float* x  = (const float*)d_in[0];   // [B,L,C]
  const float* w1 = (const float*)d_in[1];   // [C,C,3]
  const float* w2 = (const float*)d_in[2];   // [3,C,3]
  const float* wf = (const float*)d_in[3];   // [C,C,3]
  (void)in_sizes; (void)n_in; (void)out_size; (void)ws_size;

  char* ws = (char*)d_ws;
  size_t cur = 0;
  auto take = [&](size_t bytes) -> void* {
    cur = (cur + 255) & ~(size_t)255;
    void* p = ws + cur;
    cur += bytes;
    return p;
  };

  _Float16* xpadh = (_Float16*)take((size_t)DC_B * (DC_L + 2) * DC_C * 2); // 16.8 MB
  _Float16* A1h   = (_Float16*)take((size_t)DC_C * DC_KD * 2);             // 0.4 MB
  _Float16* A3h   = (_Float16*)take((size_t)DC_C * DC_KD * 2);             // 0.4 MB
  float*    obuf  = (float*)take((size_t)DC_B * DC_L * DC_C * 4);          // 33.5 MB
  float*    offb  = (float*)take((size_t)DC_B * DC_L * 3 * 4);             // 0.4 MB
  _Float16* Yh    = (_Float16*)take((size_t)DC_B * DC_L * DC_KD * 2);      // 50.3 MB

  pad_convert<<<(DC_B * (DC_L + 2) * DC_C + 255) / 256, 256, 0, stream>>>(x, xpadh);
  repack_w<<<(DC_C * DC_KD + 255) / 256, 256, 0, stream>>>(w1, A1h);
  repack_w<<<(DC_C * DC_KD + 255) / 256, 256, 0, stream>>>(wf, A3h);

  // conv1 as GEMM: B-column for l = xpad[b, l : l+3, :] (contiguous, stride C)
  gemm_f16_wmma<<<dim3(DC_L / 64, DC_C / 64, DC_B), 256, 0, stream>>>(
      A1h, xpadh, obuf, DC_C, (long long)(DC_L + 2) * DC_C, /*relu=*/1);

  conv2_offsets<<<(DC_B * DC_L + 255) / 256, 256, 0, stream>>>(obuf, w2, offb);
  interp_pack<<<DC_B * DC_L, 256, 0, stream>>>(x, offb, Yh);

  // final conv as GEMM: B-column for l = Y[b, l, :, :] (contiguous, stride 768)
  gemm_f16_wmma<<<dim3(DC_L / 64, DC_C / 64, DC_B), 256, 0, stream>>>(
      A3h, Yh, (float*)d_out, DC_KD, (long long)DC_L * DC_KD, /*relu=*/0);
}
